// GRU_Classification_56221121905314
// MI455X (gfx1250) — compile-verified, matrix operand order checked by reading, not run on hardware
//
#include <hip/hip_runtime.h>
#include <hip/hip_fp16.h>

typedef __attribute__((ext_vector_type(16))) _Float16 v16h;
typedef __attribute__((ext_vector_type(4)))  _Float16 v4h;
typedef __attribute__((ext_vector_type(8)))  float    v8f;

#define B_    64
#define T_    512
#define I_    256
#define H_    1024
#define C_    128
#define KTOT  1280        // H + I (fused [h | x_t] K dimension)
#define LDK   1288        // padded LDS row stride in halfs (conflict-free: 644 dw % 64 = 4 banks)
#define NWG   64          // H / 16 : one workgroup per 16 hidden columns
#define NTHREADS 128      // 4 waves; wave w owns batch M-tile w

union FragU { v16h v; uint4 u[2]; };

__device__ __forceinline__ void gridBarrier(unsigned* cnt, unsigned target) {
    __threadfence();
    __syncthreads();
    if (threadIdx.x == 0) {
        atomicAdd(cnt, 1u);
        while (__hip_atomic_load(cnt, __ATOMIC_ACQUIRE, __HIP_MEMORY_SCOPE_AGENT) < target) {
            __builtin_amdgcn_s_sleep(1);
        }
    }
    __syncthreads();
}

__device__ __forceinline__ void cvt_store4(_Float16* dst, const float4 s) {
    v4h h; h.x = (_Float16)s.x; h.y = (_Float16)s.y; h.z = (_Float16)s.z; h.w = (_Float16)s.w;
    *(v4h*)dst = h;
}

__global__ void __launch_bounds__(NTHREADS)
gru_persistent(const float* __restrict__ x,      // [B,T,I]
               const float* __restrict__ w_ih,   // [3H,I]
               const float* __restrict__ w_hh,   // [3H,H]
               const float* __restrict__ b_ih,   // [3H]
               const float* __restrict__ b_hh,   // [3H]
               float* __restrict__ h0buf,        // [B,H] double buffer 0
               float* __restrict__ h1buf,        // [B,H] double buffer 1
               unsigned* __restrict__ cnt)       // grid barrier counter
{
    extern __shared__ __align__(16) _Float16 smem[];
    _Float16* As = smem;                 // [64][LDK]  fused [h | x_t] in f16
    _Float16* Ws = smem + B_ * LDK;      // [48][LDK]  weight slice (r,z,n x 16 cols)

    const unsigned tid  = threadIdx.x;
    const unsigned lane = tid & 31u;
    const unsigned wv   = tid >> 5;      // wave id == M-tile (batch rows wv*16..+15)
    const unsigned g    = blockIdx.x;    // hidden-column block
    const unsigned j0   = g * 16u;

    // ---- stage this WG's weight slice into LDS once (fp32 -> f16, float4 granularity) ----
    // column c in [0,48): gate = c/16 (0=r,1=z,2=n), hidden col = j0 + c%16
    // K axis: [0,1024) = w_hh row, [1024,1280) = w_ih row. 1024 % 4 == 0, so no straddle.
    for (unsigned c = 0; c < 48u; ++c) {
        const unsigned row = (c >> 4) * H_ + j0 + (c & 15u);
        const float* wh = w_hh + (size_t)row * H_;
        const float* wi = w_ih + (size_t)row * I_;
        _Float16* dst = Ws + c * LDK;
        for (unsigned kv = tid; kv < KTOT / 4u; kv += NTHREADS) {
            unsigned k = kv * 4u;
            float4 s = (k < H_) ? *(const float4*)(wh + k)
                                : *(const float4*)(wi + (k - H_));
            cvt_store4(dst + k, s);
        }
    }

    // per-lane biases (depend only on this lane's output column)
    const unsigned jl = j0 + (lane & 15u);
    const float bR  = b_ih[jl]        + b_hh[jl];
    const float bZ  = b_ih[H_ + jl]   + b_hh[H_ + jl];
    const float bIN = b_ih[2*H_ + jl];            // stays outside reset gate
    const float bHN = b_hh[2*H_ + jl];            // multiplied by r (PyTorch semantics)

    // zero h0 columns owned by this WG (h_0 = 0)
    for (unsigned i = tid; i < B_ * 16u; i += NTHREADS) {
        h0buf[(i >> 4) * H_ + j0 + (i & 15u)] = 0.0f;
    }

    unsigned bar = 1;
    gridBarrier(cnt, NWG * bar); ++bar;

    // A-staging assignment: thread owns (row = tid/2, half = tid&1), 640 contiguous K
    const unsigned srow  = tid >> 1;
    const unsigned shalf = tid & 1u;
    const unsigned kbase = shalf * (KTOT / 2u);   // 0 or 640

    for (unsigned t = 0; t < T_; ++t) {
        float* hcur = (t & 1u) ? h1buf : h0buf;
        float* hnxt = (t & 1u) ? h0buf : h1buf;

        // ---- stage A = [h_cur | x_t] as f16 into LDS (vectorized, no div/mod) ----
        {
            const float* hrow = hcur + srow * H_;
            const float* xrow = x + (size_t)srow * (T_ * I_) + (size_t)t * I_;
            _Float16* dst = As + srow * LDK;
            if (shalf == 0) {
                // K in [0, 640): all from h
                #pragma unroll 4
                for (unsigned k = 0; k < 640u; k += 4u)
                    cvt_store4(dst + k, *(const float4*)(hrow + k));
            } else {
                // K in [640, 1024): h ; K in [1024, 1280): x_t
                #pragma unroll 4
                for (unsigned k = 640u; k < 1024u; k += 4u)
                    cvt_store4(dst + k, *(const float4*)(hrow + k));
                #pragma unroll 4
                for (unsigned k = 1024u; k < 1280u; k += 4u)
                    cvt_store4(dst + k, *(const float4*)(xrow + (k - 1024u)));
            }
            // prefetch next timestep's x slice into cache (global_prefetch_b8)
            if (t + 1u < T_)
                __builtin_prefetch(xrow + I_ + (kbase & 255u), 0, 0);
        }
        __syncthreads();

        v8f accR  = {0.f,0.f,0.f,0.f,0.f,0.f,0.f,0.f};
        v8f accZ  = accR;
        v8f accHN = accR;   // n-gate, h contribution  (K < 1024)
        v8f accXN = accR;   // n-gate, x contribution  (K >= 1024)

        const unsigned mrow = wv * 16u + (lane & 15u);
        const unsigned koff = (lane >> 4) << 3;     // lanes 16-31 start at K+8
        const _Float16* aRow = As + mrow * LDK + koff;
        const _Float16* rRow = Ws + ((lane & 15u)      ) * LDK + koff;
        const _Float16* zRow = Ws + ((lane & 15u) + 16u) * LDK + koff;
        const _Float16* nRow = Ws + ((lane & 15u) + 32u) * LDK + koff;

        // ---- K in [0, 1024): h-part (n-gate accumulates into accHN) ----
        for (unsigned k0 = 0; k0 < H_; k0 += 32u) {
            FragU a, br, bz, bn;
            a.u[0]  = *(const uint4*)(aRow + k0);
            a.u[1]  = *(const uint4*)(aRow + k0 + 16);
            br.u[0] = *(const uint4*)(rRow + k0);
            br.u[1] = *(const uint4*)(rRow + k0 + 16);
            bz.u[0] = *(const uint4*)(zRow + k0);
            bz.u[1] = *(const uint4*)(zRow + k0 + 16);
            bn.u[0] = *(const uint4*)(nRow + k0);
            bn.u[1] = *(const uint4*)(nRow + k0 + 16);
            accR  = __builtin_amdgcn_wmma_f32_16x16x32_f16(false, a.v, false, br.v, (short)0, accR,  false, false);
            accZ  = __builtin_amdgcn_wmma_f32_16x16x32_f16(false, a.v, false, bz.v, (short)0, accZ,  false, false);
            accHN = __builtin_amdgcn_wmma_f32_16x16x32_f16(false, a.v, false, bn.v, (short)0, accHN, false, false);
        }
        // ---- K in [1024, 1280): x-part (n-gate accumulates into accXN) ----
        for (unsigned k0 = H_; k0 < KTOT; k0 += 32u) {
            FragU a, br, bz, bn;
            a.u[0]  = *(const uint4*)(aRow + k0);
            a.u[1]  = *(const uint4*)(aRow + k0 + 16);
            br.u[0] = *(const uint4*)(rRow + k0);
            br.u[1] = *(const uint4*)(rRow + k0 + 16);
            bz.u[0] = *(const uint4*)(zRow + k0);
            bz.u[1] = *(const uint4*)(zRow + k0 + 16);
            bn.u[0] = *(const uint4*)(nRow + k0);
            bn.u[1] = *(const uint4*)(nRow + k0 + 16);
            accR  = __builtin_amdgcn_wmma_f32_16x16x32_f16(false, a.v, false, br.v, (short)0, accR,  false, false);
            accZ  = __builtin_amdgcn_wmma_f32_16x16x32_f16(false, a.v, false, bz.v, (short)0, accZ,  false, false);
            accXN = __builtin_amdgcn_wmma_f32_16x16x32_f16(false, a.v, false, bn.v, (short)0, accXN, false, false);
        }

        // ---- epilogue: gate nonlinearities + h update for 16 owned columns ----
        // C/D layout: VGPR v -> M = v (lanes 0-15) or v+8 (lanes 16-31), N = lane%16
        const unsigned rowbase = wv * 16u + ((lane >> 4) << 3);
        #pragma unroll
        for (int v = 0; v < 8; ++v) {
            unsigned b = rowbase + (unsigned)v;
            float r = 1.0f / (1.0f + __expf(-(accR[v] + bR)));
            float z = 1.0f / (1.0f + __expf(-(accZ[v] + bZ)));
            float n = tanhf(accXN[v] + bIN + r * (accHN[v] + bHN));
            float hprev = hcur[b * H_ + jl];
            hnxt[b * H_ + jl] = (1.0f - z) * n + z * hprev;
        }

        gridBarrier(cnt, NWG * bar); ++bar;
    }
    // T=512 (even): final hidden state ends in h0buf
}

__global__ void __launch_bounds__(C_)
fc_softmax(const float* __restrict__ h,      // [B,H] final hidden
           const float* __restrict__ fc_w,   // [C,H]
           const float* __restrict__ fc_b,   // [C]
           float* __restrict__ out)          // [B,C] softmax probs
{
    __shared__ float red[C_];
    const unsigned b = blockIdx.x, c = threadIdx.x;
    const float* hr = h + b * H_;
    const float* wr = fc_w + c * H_;
    float acc = fc_b[c];
    for (unsigned k = 0; k < H_; k += 4) {
        float4 hv = *(const float4*)(hr + k);
        float4 wv = *(const float4*)(wr + k);
        acc += hv.x * wv.x + hv.y * wv.y + hv.z * wv.z + hv.w * wv.w;
    }
    red[c] = acc;
    __syncthreads();
    for (unsigned s = C_ / 2; s > 0; s >>= 1) {
        if (c < s) red[c] = fmaxf(red[c], red[c + s]);
        __syncthreads();
    }
    float mx = red[0];
    __syncthreads();
    float e = __expf(acc - mx);
    red[c] = e;
    __syncthreads();
    for (unsigned s = C_ / 2; s > 0; s >>= 1) {
        if (c < s) red[c] += red[c + s];
        __syncthreads();
    }
    out[b * C_ + c] = e / red[0];
}

extern "C" void kernel_launch(void* const* d_in, const int* in_sizes, int n_in,
                              void* d_out, int out_size, void* d_ws, size_t ws_size,
                              hipStream_t stream) {
    (void)in_sizes; (void)n_in; (void)out_size; (void)ws_size;
    const float* x    = (const float*)d_in[0];
    const float* w_ih = (const float*)d_in[1];
    const float* w_hh = (const float*)d_in[2];
    const float* b_ih = (const float*)d_in[3];
    const float* b_hh = (const float*)d_in[4];
    const float* fc_w = (const float*)d_in[5];
    const float* fc_b = (const float*)d_in[6];
    float* out = (float*)d_out;

    // workspace: h double buffer + grid-barrier counter
    float* h0 = (float*)d_ws;
    float* h1 = h0 + B_ * H_;
    unsigned* cnt = (unsigned*)(h1 + B_ * H_);

    hipMemsetAsync(cnt, 0, sizeof(unsigned), stream);   // graph-capturable

    const size_t lds = (size_t)(B_ + 48) * LDK * sizeof(_Float16);  // ~282 KB < 320 KB/WGP
    hipFuncSetAttribute((const void*)gru_persistent,
                        hipFuncAttributeMaxDynamicSharedMemorySize, (int)lds);

    gru_persistent<<<NWG, NTHREADS, lds, stream>>>(x, w_ih, w_hh, b_ih, b_hh, h0, h1, cnt);
    fc_softmax<<<B_, C_, 0, stream>>>(h0, fc_w, fc_b, out);
}
